// GSN_42709154791890
// MI455X (gfx1250) — matrix operand order repaired and checked by reading, not compiled, not run on hardware
//
#include <hip/hip_runtime.h>
#include <hip/hip_bf16.h>

// ---------------------------------------------------------------------------
// Problem constants (from the reference)
// ---------------------------------------------------------------------------
#define N_NODES   20000
#define N_EDGES   160000
#define IN_DIM    64
#define HID       256
#define EDGE_DIM  64
#define N_LAYERS  3
#define NUM_GRAPHS 128
#define MSG_K     576     // 2*HID + EDGE_DIM (sf columns handled as rank-1)
#define UPD_K     512
#define ROWS      32      // rows (edges/nodes) per block: two 16-row WMMA tiles

typedef __attribute__((ext_vector_type(16))) __bf16 v16bf;
typedef __attribute__((ext_vector_type(8)))  __bf16 v8bf;
typedef __attribute__((ext_vector_type(8)))  float  v8f;
typedef __attribute__((ext_vector_type(4)))  float  v4f;

// ---------------------------------------------------------------------------
// WMMA helpers
// ---------------------------------------------------------------------------
__device__ __forceinline__ v8f wmma_bf16(v16bf a, v16bf b, v8f c) {
  return __builtin_amdgcn_wmma_f32_16x16x32_bf16(
      /*neg_a=*/false, a, /*neg_b=*/false, b,
      /*c_mod=*/(short)0, c, /*reuse_a=*/false, /*reuse_b=*/false);
}

// A fragment (16 rows x 32 K) from a row-major bf16 matrix in LDS.
// ISA 16-bit A layout: lane<16 -> row=lane,   K = kbase+{0..7}, kbase+16+{0..7}
//                      lane>=16 -> row=lane-16, K = kbase+8+{0..7}, kbase+24+{0..7}
__device__ __forceinline__ v16bf load_a_frag(const __bf16* base, int ldh,
                                             int kbase, int lane) {
  const int row  = lane & 15;
  const int koff = (lane >> 4) << 3;
  const __bf16* p = base + row * ldh + kbase + koff;
  v8bf lo = *(const v8bf*)(p);
  v8bf hi = *(const v8bf*)(p + 16);
  v16bf r;
#pragma unroll
  for (int i = 0; i < 8; ++i) { r[i] = lo[i]; r[i + 8] = hi[i]; }
  return r;
}

// B fragment from pre-packed weights: frag = kt*(N/16)+nt, 512 halfs/frag,
// lane-major (32 lanes x 16 halfs = contiguous 32B per lane).
__device__ __forceinline__ v16bf load_b_frag(const __bf16* wp, int frag,
                                             int lane) {
  return *(const v16bf*)(wp + ((size_t)frag << 9) + (lane << 4));
}

// 16-byte staging helpers
__device__ __forceinline__ void stage_bf16_8(const __bf16* src, __bf16* dst) {
  *(v8bf*)dst = *(const v8bf*)src;
}
__device__ __forceinline__ void stage_f32_to_bf16_8(const float* src,
                                                    __bf16* dst) {
  v4f a = *(const v4f*)src;
  v4f b = *(const v4f*)(src + 4);
  v8bf o;
#pragma unroll
  for (int i = 0; i < 4; ++i) { o[i] = (__bf16)a[i]; o[i + 4] = (__bf16)b[i]; }
  *(v8bf*)dst = o;
}

// ---------------------------------------------------------------------------
// Weight packing: f32 [K x N] row-major -> bf16 WMMA B-fragment order.
// B layout (16-bit): lane l holds col n = nt*16+(l&15);
//   half i (0..15) holds k = kt*32 + (l>=16 ? 8 : 0) + (i&7) + (i>=8 ? 16 : 0).
// skip_at >= 0: skip 2 rows at k==skip_at (msg_W1 sf rows handled separately).
// ---------------------------------------------------------------------------
__global__ void pack_w_kernel(const float* __restrict__ W, int Kp, int N,
                              int skip_at, __bf16* __restrict__ out) {
  int i = blockIdx.x * blockDim.x + threadIdx.x;
  int total = Kp * N;
  if (i >= total) return;
  int f    = i >> 9;
  int j    = i & 511;
  int lane = j >> 4;
  int ii   = j & 15;
  int ntiles = N >> 4;
  int nt = f % ntiles;
  int kt = f / ntiles;
  int n  = nt * 16 + (lane & 15);
  int kk = ((lane >> 4) << 3) + (ii & 7) + ((ii >> 3) << 4);
  int kp = kt * 32 + kk;
  int ko = kp + ((skip_at >= 0 && kp >= skip_at) ? 2 : 0);
  out[i] = (__bf16)W[(size_t)ko * N + n];
}

__global__ void zero_f32_kernel(float* __restrict__ p, int n) {
  int i = blockIdx.x * blockDim.x + threadIdx.x;
  if (i < n) p[i] = 0.f;
}

// ---------------------------------------------------------------------------
// h = x @ Wlin + blin  -> h (f32) and h_bf (bf16). 32 rows/block.
// ---------------------------------------------------------------------------
__global__ __launch_bounds__(256)
void lin_kernel(const float* __restrict__ x, const float* __restrict__ blin,
                const __bf16* __restrict__ Wp,
                float* __restrict__ h, __bf16* __restrict__ hbf) {
  __shared__ __bf16 Alds[ROWS * IN_DIM];
  const int tid = threadIdx.x;
  const int r0  = blockIdx.x * ROWS;

  for (int c = tid; c < ROWS * IN_DIM / 8; c += 256)
    stage_f32_to_bf16_8(x + (size_t)r0 * IN_DIM + c * 8, Alds + c * 8);
  __syncthreads();

  const int wave = tid >> 5, lane = tid & 31;
  const int nt0 = wave * 2, nt1 = nt0 + 1;

  v8f acc00 = {}, acc01 = {}, acc10 = {}, acc11 = {};
#pragma unroll
  for (int kt = 0; kt < IN_DIM / 32; ++kt) {
    v16bf a0 = load_a_frag(Alds, IN_DIM, kt * 32, lane);
    v16bf a1 = load_a_frag(Alds + 16 * IN_DIM, IN_DIM, kt * 32, lane);
    v16bf b0 = load_b_frag(Wp, kt * (HID / 16) + nt0, lane);
    v16bf b1 = load_b_frag(Wp, kt * (HID / 16) + nt1, lane);
    acc00 = wmma_bf16(a0, b0, acc00);
    acc10 = wmma_bf16(a1, b0, acc10);
    acc01 = wmma_bf16(a0, b1, acc01);
    acc11 = wmma_bf16(a1, b1, acc11);
  }
  const int mbase = (lane >> 4) * 8;
#pragma unroll
  for (int t = 0; t < 2; ++t) {
    const v8f& c0 = t ? acc01 : acc00;
    const v8f& c1 = t ? acc11 : acc10;
    const int n = (nt0 + t) * 16 + (lane & 15);
    const float bias = blin[n];
#pragma unroll
    for (int r = 0; r < 8; ++r) {
      const int m0 = mbase + r, m1 = 16 + mbase + r;
      float v0 = c0[r] + bias;
      float v1 = c1[r] + bias;
      h[(size_t)(r0 + m0) * HID + n]   = v0;
      h[(size_t)(r0 + m1) * HID + n]   = v1;
      hbf[(size_t)(r0 + m0) * HID + n] = (__bf16)v0;
      hbf[(size_t)(r0 + m1) * HID + n] = (__bf16)v1;
    }
  }
}

// ---------------------------------------------------------------------------
// Edge message MLP + weighted scatter into upd. 32 edges/block.
// ---------------------------------------------------------------------------
__global__ __launch_bounds__(256)
void msg_kernel(const __bf16* __restrict__ hbf,
                const float* __restrict__ node_sf,
                const float* __restrict__ edge_feature,
                const float* __restrict__ edge_weight,
                const int*   __restrict__ edge_list,
                const float* __restrict__ W1_full,  // 578x256 f32 (sf rows)
                const float* __restrict__ b1,
                const float* __restrict__ b2,
                const __bf16* __restrict__ W1p,     // 18x16 frags
                const __bf16* __restrict__ W2p,     // 8x16 frags
                float* __restrict__ upd) {
  __shared__ __bf16 Alds[ROWS * MSG_K];   // 36 KB
  __shared__ __bf16 Hlds[ROWS * HID];     // 16 KB
  __shared__ int   s_nin[ROWS], s_nout[ROWS];
  __shared__ float s_ew[ROWS], s_sfi[ROWS], s_sfo[ROWS];

  const int tid = threadIdx.x;
  const int e0  = blockIdx.x * ROWS;

  if (tid < ROWS) {
    const int e  = e0 + tid;
    const int ni = edge_list[2 * e + 0];
    const int no = edge_list[2 * e + 1];
    s_nin[tid]  = ni;
    s_nout[tid] = no;
    s_ew[tid]   = edge_weight[e];
    s_sfi[tid]  = node_sf[ni];
    s_sfo[tid]  = node_sf[no];
  }
  __syncthreads();

  // stage A in 8-half (16B) chunks: 72 chunks per row
  for (int c = tid; c < ROWS * (MSG_K / 8); c += 256) {
    const int row = c / (MSG_K / 8);
    const int j   = c - row * (MSG_K / 8);
    __bf16* dst = Alds + row * MSG_K + j * 8;
    if (j < 32)
      stage_bf16_8(hbf + (size_t)s_nin[row] * HID + j * 8, dst);
    else if (j < 64)
      stage_bf16_8(hbf + (size_t)s_nout[row] * HID + (j - 32) * 8, dst);
    else
      stage_f32_to_bf16_8(
          edge_feature + (size_t)(e0 + row) * EDGE_DIM + (j - 64) * 8, dst);
  }
  __syncthreads();

  const int wave = tid >> 5, lane = tid & 31;
  const int nt0 = wave * 2, nt1 = nt0 + 1;
  const int mbase = (lane >> 4) * 8;

  // ---- hidden = relu(m @ W1 + b1 + sf rank-1) ----
  {
    v8f acc00 = {}, acc01 = {}, acc10 = {}, acc11 = {};
#pragma unroll
    for (int kt = 0; kt < MSG_K / 32; ++kt) {
      v16bf a0 = load_a_frag(Alds, MSG_K, kt * 32, lane);
      v16bf a1 = load_a_frag(Alds + 16 * MSG_K, MSG_K, kt * 32, lane);
      v16bf b0 = load_b_frag(W1p, kt * (HID / 16) + nt0, lane);
      v16bf b1 = load_b_frag(W1p, kt * (HID / 16) + nt1, lane);
      acc00 = wmma_bf16(a0, b0, acc00);
      acc10 = wmma_bf16(a1, b0, acc10);
      acc01 = wmma_bf16(a0, b1, acc01);
      acc11 = wmma_bf16(a1, b1, acc11);
    }
#pragma unroll
    for (int t = 0; t < 2; ++t) {
      const v8f& c0 = t ? acc01 : acc00;
      const v8f& c1 = t ? acc11 : acc10;
      const int n = (nt0 + t) * 16 + (lane & 15);
      const float bias = b1[n];
      const float w512 = W1_full[512 * HID + n];
      const float w513 = W1_full[513 * HID + n];
#pragma unroll
      for (int r = 0; r < 8; ++r) {
        const int m0 = mbase + r, m1 = 16 + mbase + r;
        float v0 = c0[r] + bias + s_sfi[m0] * w512 + s_sfo[m0] * w513;
        float v1 = c1[r] + bias + s_sfi[m1] * w512 + s_sfo[m1] * w513;
        Hlds[m0 * HID + n] = (__bf16)(v0 > 0.f ? v0 : 0.f);
        Hlds[m1 * HID + n] = (__bf16)(v1 > 0.f ? v1 : 0.f);
      }
    }
  }
  __syncthreads();

  // ---- out = hidden @ W2 + b2 ; scatter (out * ew) into upd[node_out] ----
  {
    v8f acc00 = {}, acc01 = {}, acc10 = {}, acc11 = {};
#pragma unroll
    for (int kt = 0; kt < HID / 32; ++kt) {
      v16bf a0 = load_a_frag(Hlds, HID, kt * 32, lane);
      v16bf a1 = load_a_frag(Hlds + 16 * HID, HID, kt * 32, lane);
      v16bf b0 = load_b_frag(W2p, kt * (HID / 16) + nt0, lane);
      v16bf b1 = load_b_frag(W2p, kt * (HID / 16) + nt1, lane);
      acc00 = wmma_bf16(a0, b0, acc00);
      acc10 = wmma_bf16(a1, b0, acc10);
      acc01 = wmma_bf16(a0, b1, acc01);
      acc11 = wmma_bf16(a1, b1, acc11);
    }
#pragma unroll
    for (int t = 0; t < 2; ++t) {
      const v8f& c0 = t ? acc01 : acc00;
      const v8f& c1 = t ? acc11 : acc10;
      const int n = (nt0 + t) * 16 + (lane & 15);
      const float bias = b2[n];
#pragma unroll
      for (int r = 0; r < 8; ++r) {
        const int m0 = mbase + r, m1 = 16 + mbase + r;
        atomicAdd(&upd[(size_t)s_nout[m0] * HID + n], (c0[r] + bias) * s_ew[m0]);
        atomicAdd(&upd[(size_t)s_nout[m1] * HID + n], (c1[r] + bias) * s_ew[m1]);
      }
    }
  }
}

// ---------------------------------------------------------------------------
// Node update MLP: c=[h|upd], h = relu(relu(c@U1+b1)@U2+b2). 32 nodes/block.
// ---------------------------------------------------------------------------
__global__ __launch_bounds__(256)
void upd_kernel(float* __restrict__ h, __bf16* __restrict__ hbf,
                const float* __restrict__ updv,
                const float* __restrict__ b1, const float* __restrict__ b2,
                const __bf16* __restrict__ U1p, const __bf16* __restrict__ U2p) {
  __shared__ __bf16 Alds[ROWS * UPD_K];   // 32 KB
  __shared__ __bf16 Hlds[ROWS * HID];     // 16 KB
  const int tid = threadIdx.x;
  const int r0  = blockIdx.x * ROWS;

  for (int c = tid; c < ROWS * (UPD_K / 8); c += 256) {
    const int row = c >> 6;
    const int j   = c & 63;
    __bf16* dst = Alds + row * UPD_K + j * 8;
    if (j < 32)
      stage_bf16_8(hbf + (size_t)(r0 + row) * HID + j * 8, dst);
    else
      stage_f32_to_bf16_8(updv + (size_t)(r0 + row) * HID + (j - 32) * 8, dst);
  }
  __syncthreads();

  const int wave = tid >> 5, lane = tid & 31;
  const int nt0 = wave * 2, nt1 = nt0 + 1;
  const int mbase = (lane >> 4) * 8;

  {
    v8f acc00 = {}, acc01 = {}, acc10 = {}, acc11 = {};
#pragma unroll
    for (int kt = 0; kt < UPD_K / 32; ++kt) {
      v16bf a0 = load_a_frag(Alds, UPD_K, kt * 32, lane);
      v16bf a1 = load_a_frag(Alds + 16 * UPD_K, UPD_K, kt * 32, lane);
      v16bf b0 = load_b_frag(U1p, kt * (HID / 16) + nt0, lane);
      v16bf b1 = load_b_frag(U1p, kt * (HID / 16) + nt1, lane);
      acc00 = wmma_bf16(a0, b0, acc00);
      acc10 = wmma_bf16(a1, b0, acc10);
      acc01 = wmma_bf16(a0, b1, acc01);
      acc11 = wmma_bf16(a1, b1, acc11);
    }
#pragma unroll
    for (int t = 0; t < 2; ++t) {
      const v8f& c0 = t ? acc01 : acc00;
      const v8f& c1 = t ? acc11 : acc10;
      const int n = (nt0 + t) * 16 + (lane & 15);
      const float bias = b1[n];
#pragma unroll
      for (int r = 0; r < 8; ++r) {
        const int m0 = mbase + r, m1 = 16 + mbase + r;
        float v0 = c0[r] + bias;
        float v1 = c1[r] + bias;
        Hlds[m0 * HID + n] = (__bf16)(v0 > 0.f ? v0 : 0.f);
        Hlds[m1 * HID + n] = (__bf16)(v1 > 0.f ? v1 : 0.f);
      }
    }
  }
  __syncthreads();

  {
    v8f acc00 = {}, acc01 = {}, acc10 = {}, acc11 = {};
#pragma unroll
    for (int kt = 0; kt < HID / 32; ++kt) {
      v16bf a0 = load_a_frag(Hlds, HID, kt * 32, lane);
      v16bf a1 = load_a_frag(Hlds + 16 * HID, HID, kt * 32, lane);
      v16bf b0 = load_b_frag(U2p, kt * (HID / 16) + nt0, lane);
      v16bf b1 = load_b_frag(U2p, kt * (HID / 16) + nt1, lane);
      acc00 = wmma_bf16(a0, b0, acc00);
      acc10 = wmma_bf16(a1, b0, acc10);
      acc01 = wmma_bf16(a0, b1, acc01);
      acc11 = wmma_bf16(a1, b1, acc11);
    }
#pragma unroll
    for (int t = 0; t < 2; ++t) {
      const v8f& c0 = t ? acc01 : acc00;
      const v8f& c1 = t ? acc11 : acc10;
      const int n = (nt0 + t) * 16 + (lane & 15);
      const float bias = b2[n];
#pragma unroll
      for (int r = 0; r < 8; ++r) {
        const int m0 = mbase + r, m1 = 16 + mbase + r;
        float v0 = c0[r] + bias; v0 = v0 > 0.f ? v0 : 0.f;   // final ReLU
        float v1 = c1[r] + bias; v1 = v1 > 0.f ? v1 : 0.f;
        h[(size_t)(r0 + m0) * HID + n]   = v0;
        h[(size_t)(r0 + m1) * HID + n]   = v1;
        hbf[(size_t)(r0 + m0) * HID + n] = (__bf16)v0;
        hbf[(size_t)(r0 + m1) * HID + n] = (__bf16)v1;
      }
    }
  }
}

// ---------------------------------------------------------------------------
// Graph pooling: out[g] = sum_{node2graph[i]==g} h[i]
// ---------------------------------------------------------------------------
__global__ void pool_kernel(const float* __restrict__ h,
                            const int* __restrict__ n2g,
                            float* __restrict__ out) {
  int i = blockIdx.x * blockDim.x + threadIdx.x;
  if (i < N_NODES * HID) {
    const int node = i >> 8;
    const int c    = i & 255;
    atomicAdd(&out[n2g[node] * HID + c], h[i]);
  }
}

// ---------------------------------------------------------------------------
// Launch
// ---------------------------------------------------------------------------
extern "C" void kernel_launch(void* const* d_in, const int* in_sizes, int n_in,
                              void* d_out, int out_size, void* d_ws, size_t ws_size,
                              hipStream_t stream) {
  (void)in_sizes; (void)n_in; (void)out_size; (void)ws_size;

  const float* x    = (const float*)d_in[0];
  const float* nsf  = (const float*)d_in[1];
  const float* ef   = (const float*)d_in[2];
  const float* ew   = (const float*)d_in[3];
  const int*   el   = (const int*)d_in[4];
  const int*   n2g  = (const int*)d_in[5];
  const float* Wlin = (const float*)d_in[6];
  const float* blin = (const float*)d_in[7];
  const float* mW1  = (const float*)d_in[8];
  const float* mb1  = (const float*)d_in[9];
  const float* mW2  = (const float*)d_in[10];
  const float* mb2  = (const float*)d_in[11];
  const float* uW1  = (const float*)d_in[12];
  const float* ub1  = (const float*)d_in[13];
  const float* uW2  = (const float*)d_in[14];
  const float* ub2  = (const float*)d_in[15];
  float* out = (float*)d_out;

  // ---- workspace layout ----
  char* ws = (char*)d_ws;
  float*  h    = (float*)(ws);                       // 20000*256 f32
  __bf16* hbf  = (__bf16*)(ws + 20480000);           // 20000*256 bf16
  float*  updv = (float*)(ws + 30720000);            // 20000*256 f32
  __bf16* wp   = (__bf16*)(ws + 51200000);           // packed weights

  __bf16* wlin_p = wp;                               // 16384 halfs
  const size_t L_HALFS = 409600;                     // per-layer packed halfs
  auto layer_base = [&](int l) { return wp + 16384 + (size_t)l * L_HALFS; };
  // per-layer offsets (halfs): mW1p 0, mW2p 147456, uW1p 212992, uW2p 344064
  const dim3 B(256);

  // ---- pack weights -> bf16 fragment order ----
  pack_w_kernel<<<(16384 + 255) / 256, B, 0, stream>>>(Wlin, 64, HID, -1, wlin_p);
  for (int l = 0; l < N_LAYERS; ++l) {
    __bf16* base = layer_base(l);
    pack_w_kernel<<<(147456 + 255) / 256, B, 0, stream>>>(
        mW1 + (size_t)l * 578 * HID, MSG_K, HID, 512, base);
    pack_w_kernel<<<(65536 + 255) / 256, B, 0, stream>>>(
        mW2 + (size_t)l * HID * HID, HID, HID, -1, base + 147456);
    pack_w_kernel<<<(131072 + 255) / 256, B, 0, stream>>>(
        uW1 + (size_t)l * UPD_K * HID, UPD_K, HID, -1, base + 212992);
    pack_w_kernel<<<(65536 + 255) / 256, B, 0, stream>>>(
        uW2 + (size_t)l * HID * HID, HID, HID, -1, base + 344064);
  }

  // ---- input linear ----
  lin_kernel<<<N_NODES / ROWS, B, 0, stream>>>(x, blin, wlin_p, h, hbf);

  // ---- message-passing layers ----
  for (int l = 0; l < N_LAYERS; ++l) {
    __bf16* base = layer_base(l);
    zero_f32_kernel<<<(N_NODES * HID + 255) / 256, B, 0, stream>>>(
        updv, N_NODES * HID);
    msg_kernel<<<N_EDGES / ROWS, B, 0, stream>>>(
        hbf, nsf, ef, ew, el,
        mW1 + (size_t)l * 578 * HID, mb1 + (size_t)l * HID,
        mb2 + (size_t)l * HID, base, base + 147456, updv);
    upd_kernel<<<N_NODES / ROWS, B, 0, stream>>>(
        h, hbf, updv, ub1 + (size_t)l * HID, ub2 + (size_t)l * HID,
        base + 212992, base + 344064);
  }

  // ---- graph pooling ----
  zero_f32_kernel<<<(NUM_GRAPHS * HID + 255) / 256, B, 0, stream>>>(
      out, NUM_GRAPHS * HID);
  pool_kernel<<<(N_NODES * HID + 255) / 256, B, 0, stream>>>(h, n2g, out);
}
